// Dual_axis_3341484556966
// MI455X (gfx1250) — matrix operand order repaired and checked by reading, not compiled, not run on hardware
//
#include <hip/hip_runtime.h>
#include <hip/hip_bf16.h>

// ---------------------------------------------------------------------------
// Dual-axis attention for MI455X (gfx1250, wave32, WMMA 16x16x32 f16).
// - All GEMMs on v_wmma_f32_16x16x32_f16 (fp32 accumulate)
// - LDS tiles are f16 in fragment-chunk order: every fragment = 2x ds_load_b128
// - f16 intermediates everywhere (vconv, vlin, rbuf, Rh) -> minimal HBM traffic
// ---------------------------------------------------------------------------

typedef __attribute__((ext_vector_type(16))) _Float16 v16h;
typedef __attribute__((ext_vector_type(8)))  _Float16 v8h;
typedef __attribute__((ext_vector_type(2)))  _Float16 v2h;
typedef __attribute__((ext_vector_type(8)))  float    v8f;

#define CCONST 256

__device__ __forceinline__ v8f wmma_f16(v16h a, v16h b, v8f c) {
  return __builtin_amdgcn_wmma_f32_16x16x32_f16(false, a, false, b, (short)0, c,
                                                false, false);
}

__device__ __forceinline__ v16h cat8(v8h a, v8h b) {
  return __builtin_shufflevector(a, b, 0, 1, 2, 3, 4, 5, 6, 7,
                                 8, 9, 10, 11, 12, 13, 14, 15);
}

// A fragment (16x32 f16): lane m = base+(lane&15), h = lane>>4.
// element j -> K = (j&7) + 8h + 16*(j>>3): two contiguous 8-half chunks.
__device__ __forceinline__ v16h afrag(const _Float16* row, int h) {
  return cat8(*(const v8h*)(row + 8 * h), *(const v8h*)(row + 16 + 8 * h));
}
// B fragment (32x16 f16): lane n = base+(lane&15); element j -> K = (j&7) + 8h + 16*(j>>3).
__device__ __forceinline__ v16h bfrag(const _Float16* row, int h) {
  return cat8(*(const v8h*)(row + 8 * h), *(const v8h*)(row + 16 + 8 * h));
}

// 16B-chunk loaders for global->LDS staging (k-contiguous sources)
__device__ __forceinline__ v8h load8h(const float* p) {
  v8h r;
#pragma unroll
  for (int j = 0; j < 8; ++j) r[j] = (_Float16)p[j];
  return r;
}
__device__ __forceinline__ v8h load8h(const _Float16* p) { return *(const v8h*)p; }
// pair loaders for transpose staging
__device__ __forceinline__ v2h load2h(const float* p) {
  v2h r; r[0] = (_Float16)p[0]; r[1] = (_Float16)p[1]; return r;
}
__device__ __forceinline__ v2h load2h(const _Float16* p) { return *(const v2h*)p; }

// ---------------------------------------------------------------------------
// K1: fused 5x depthwise 3x3 conv. One block per (b,c) plane.
// Each thread handles 8 horizontal pixel-pairs -> packed v2h f16 stores.
// ---------------------------------------------------------------------------
__global__ __launch_bounds__(256) void dwconv_pool_kernel(
    const float* __restrict__ x,
    const float* __restrict__ wqh, const float* __restrict__ bqh,
    const float* __restrict__ wkh, const float* __restrict__ bkh,
    const float* __restrict__ wv,  const float* __restrict__ bv,
    const float* __restrict__ wqw, const float* __restrict__ bqw,
    const float* __restrict__ wkw, const float* __restrict__ bkw,
    _Float16* __restrict__ vconv,
    float* __restrict__ qh_pool, float* __restrict__ kh_pool,
    float* __restrict__ qw_pool, float* __restrict__ kw_pool) {
  __shared__ float tile[66 * 68];
  __shared__ float rs_qh[64], rs_kh[64], cs_qw[64], cs_kw[64];

  const int tid = threadIdx.x;
  const int b = blockIdx.x >> 8;
  const int c = blockIdx.x & 255;
  const size_t plane = ((size_t)b * CCONST + c) << 12;

  if (tid < 64) { rs_qh[tid] = 0.f; rs_kh[tid] = 0.f; cs_qw[tid] = 0.f; cs_kw[tid] = 0.f; }

  for (int idx = tid; idx < 66 * 66; idx += 256) {
    int r = idx / 66, cc = idx % 66;
    int gr = r - 1, gc = cc - 1;
    float v = 0.f;
    if (gr >= 0 && gr < 64 && gc >= 0 && gc < 64) v = x[plane + gr * 64 + gc];
    tile[r * 68 + cc] = v;
  }

  float w0[9], w1[9], w2[9], w3[9], w4[9];
#pragma unroll
  for (int t = 0; t < 9; ++t) {
    w0[t] = wqh[c * 9 + t]; w1[t] = wkh[c * 9 + t]; w2[t] = wv[c * 9 + t];
    w3[t] = wqw[c * 9 + t]; w4[t] = wkw[c * 9 + t];
  }
  const float b0 = bqh[c], b1 = bkh[c], b2 = bv[c], b3 = bqw[c], b4 = bkw[c];

  __syncthreads();

  const int c0 = 2 * (tid & 31);             // fixed column pair per thread
  float qw0 = 0.f, qw1 = 0.f, kw0 = 0.f, kw1 = 0.f;
#pragma unroll
  for (int i = 0; i < 8; ++i) {
    int q = tid + 256 * i;                   // pixel-pair index
    int r = q >> 5;
    float s0[2], s1[2], s2[2], s3[2], s4[2];
#pragma unroll
    for (int e = 0; e < 2; ++e) {
      float a0 = b0, a1 = b1, a2 = b2, a3 = b3, a4 = b4;
#pragma unroll
      for (int dy = 0; dy < 3; ++dy)
#pragma unroll
        for (int dx = 0; dx < 3; ++dx) {
          float xv = tile[(r + dy) * 68 + (c0 + e + dx)];
          int t = dy * 3 + dx;
          a0 += w0[t] * xv; a1 += w1[t] * xv; a2 += w2[t] * xv;
          a3 += w3[t] * xv; a4 += w4[t] * xv;
        }
      s0[e] = a0; s1[e] = a1; s2[e] = a2; s3[e] = a3; s4[e] = a4;
    }
    v2h vo; vo[0] = (_Float16)s2[0]; vo[1] = (_Float16)s2[1];
    *(v2h*)&vconv[plane + 2 * q] = vo;       // f16 v-conv, [b][c][pos]
    atomicAdd(&rs_qh[r], s0[0] + s0[1]);
    atomicAdd(&rs_kh[r], s1[0] + s1[1]);
    qw0 += s3[0]; qw1 += s3[1];
    kw0 += s4[0]; kw1 += s4[1];
  }
  atomicAdd(&cs_qw[c0], qw0); atomicAdd(&cs_qw[c0 + 1], qw1);
  atomicAdd(&cs_kw[c0], kw0); atomicAdd(&cs_kw[c0 + 1], kw1);
  __syncthreads();

  if (tid < 64) {
    const float inv = 1.0f / 64.0f;
    size_t o = ((size_t)b * 64 + tid) * CCONST + c;
    qh_pool[o] = rs_qh[tid] * inv;
    kh_pool[o] = rs_kh[tid] * inv;
    qw_pool[o] = cs_qw[tid] * inv;
    kw_pool[o] = cs_kw[tid] * inv;
  }
}

// ---------------------------------------------------------------------------
// Generic WMMA GEMM: out[M,N] = A[M,K] . W[N,K]^T + bias (torch Linear).
// BM=128, BN=64, BK=32; 8 waves; each wave: 1 A-frag reused over 4 WMMAs.
// ---------------------------------------------------------------------------
template <typename AT, typename OT>
__global__ __launch_bounds__(256) void gemm_lin_kernel(
    const AT* __restrict__ A, const float* __restrict__ W,
    const float* __restrict__ bias, OT* __restrict__ out,
    int M, int N, int K) {
  __shared__ __align__(16) _Float16 As[128 * 40];
  __shared__ __align__(16) _Float16 Bs[64 * 40];
  const int tid = threadIdx.x, lane = tid & 31, wv = tid >> 5;
  const int m0 = blockIdx.y * 128, n0 = blockIdx.x * 64;
  const int tm = wv * 16;
  const int h = lane >> 4, l15 = lane & 15;
  v8f acc[4] = {{}, {}, {}, {}};
  (void)M;

  for (int kk = 0; kk < K; kk += 32) {
#pragma unroll
    for (int i = 0; i < 2; ++i) {            // A: 128x32 = 512 x 8-half chunks
      int p = tid + 256 * i;
      int r = p >> 2, co = p & 3;
      *(v8h*)&As[r * 40 + 8 * co] = load8h(&A[(size_t)(m0 + r) * K + kk + 8 * co]);
    }
    {                                        // B: 64x32 = 256 chunks
      int r = tid >> 2, co = tid & 3;
      *(v8h*)&Bs[r * 40 + 8 * co] = load8h(&W[(size_t)(n0 + r) * K + kk + 8 * co]);
    }
    if (kk + 32 < K) {                       // prefetch next k-tile
      __builtin_prefetch(&A[(size_t)(m0 + (tid >> 1)) * K + kk + 32], 0, 1);
      __builtin_prefetch(&W[(size_t)(n0 + (tid >> 2)) * K + kk + 32], 0, 1);
    }
    __syncthreads();
    v16h af = afrag(&As[(tm + l15) * 40], h);
#pragma unroll
    for (int t = 0; t < 4; ++t) {
      v16h bf = bfrag(&Bs[(16 * t + l15) * 40], h);
      acc[t] = wmma_f16(af, bf, acc[t]);
    }
    __syncthreads();
  }
#pragma unroll
  for (int t = 0; t < 4; ++t) {
    int n = n0 + 16 * t + l15;
    float bi = bias[n];
#pragma unroll
    for (int v = 0; v < 8; ++v) {
      int m = m0 + tm + v + 8 * h;
      out[(size_t)m * N + n] = (OT)(acc[t][v] + bi);
    }
  }
}

// ---------------------------------------------------------------------------
// fc_v GEMM, batched transposed-A: A[m=b*4096+pos, k=c] at
// vconv[(b*256+c)*4096 + pos] (f16). M=65536, N=256, K=256, out f16.
// ---------------------------------------------------------------------------
__global__ __launch_bounds__(256) void gemm_fcv_kernel(
    const _Float16* __restrict__ vconv, const float* __restrict__ W,
    const float* __restrict__ bias, _Float16* __restrict__ out) {
  __shared__ __align__(16) _Float16 As[128 * 40];
  __shared__ __align__(16) _Float16 Bs[64 * 40];
  const int tid = threadIdx.x, lane = tid & 31, wv = tid >> 5;
  const int m0 = blockIdx.y * 128, n0 = blockIdx.x * 64;
  const int tm = wv * 16;
  const int h = lane >> 4, l15 = lane & 15;
  const int b = m0 >> 12;                    // block fits within one batch
  const size_t basev = (size_t)b * CCONST * 4096 + (size_t)(m0 & 4095);
  v8f acc[4] = {{}, {}, {}, {}};

  for (int kk = 0; kk < 256; kk += 32) {
#pragma unroll
    for (int i = 0; i < 8; ++i) {            // A: pairs along pos (coalesced)
      int p = tid + 256 * i;
      int k = p >> 6, mp = p & 63;
      v2h g = *(const v2h*)&vconv[basev + (size_t)(kk + k) * 4096 + 2 * mp];
      As[(2 * mp) * 40 + k]     = g[0];
      As[(2 * mp + 1) * 40 + k] = g[1];
    }
    {
      int r = tid >> 2, co = tid & 3;
      *(v8h*)&Bs[r * 40 + 8 * co] = load8h(&W[(size_t)(n0 + r) * 256 + kk + 8 * co]);
    }
    if (kk + 32 < 256)
      __builtin_prefetch(&vconv[basev + (size_t)(kk + 32 + (tid >> 3)) * 4096], 0, 1);
    __syncthreads();
    v16h af = afrag(&As[(tm + l15) * 40], h);
#pragma unroll
    for (int t = 0; t < 4; ++t) {
      v16h bf = bfrag(&Bs[(16 * t + l15) * 40], h);
      acc[t] = wmma_f16(af, bf, acc[t]);
    }
    __syncthreads();
  }
#pragma unroll
  for (int t = 0; t < 4; ++t) {
    int n = n0 + 16 * t + l15;
    float bi = bias[n];
#pragma unroll
    for (int v = 0; v < 8; ++v) {
      int m = m0 + tm + v + 8 * h;
      out[(size_t)m * 256 + n] = (_Float16)(acc[t][v] + bi);
    }
  }
}

// ---------------------------------------------------------------------------
// Attention scores + softmax, one block per (b, head).
// ---------------------------------------------------------------------------
__global__ __launch_bounds__(256) void attn_kernel(
    const float* __restrict__ qlin, const float* __restrict__ klin,
    const float* __restrict__ bias2d, float* __restrict__ attn, float scale) {
  __shared__ __align__(16) _Float16 Qs[64 * 40];
  __shared__ __align__(16) _Float16 Ks[64 * 40];
  __shared__ float Ss[64][65];
  const int tid = threadIdx.x, lane = tid & 31, wv = tid >> 5;
  const int bH = blockIdx.x, b = bH >> 3, Hh = bH & 7;
  const int tm = (wv & 3) * 16, tn = (wv >> 2) * 32;
  const int h = lane >> 4, l15 = lane & 15;

  {                                          // 64x32 each = 256 chunks each
    int r = tid >> 2, co = tid & 3;
    size_t o = ((size_t)b * 64 + r) * 256 + Hh * 32 + 8 * co;
    *(v8h*)&Qs[r * 40 + 8 * co] = load8h(&qlin[o]);
    *(v8h*)&Ks[r * 40 + 8 * co] = load8h(&klin[o]);
  }
  __syncthreads();

  v8f acc0 = {}, acc1 = {};
  v16h af  = afrag(&Qs[(tm + l15) * 40], h);
  v16h bf0 = bfrag(&Ks[(tn + l15) * 40], h);
  v16h bf1 = bfrag(&Ks[(tn + 16 + l15) * 40], h);
  acc0 = wmma_f16(af, bf0, acc0);
  acc1 = wmma_f16(af, bf1, acc1);

  const int nA = tn + l15, nB = nA + 16;
#pragma unroll
  for (int v = 0; v < 8; ++v) {
    int m = tm + v + 8 * h;
    Ss[m][nA] = acc0[v] * scale + bias2d[((size_t)Hh * 64 + m) * 64 + nA];
    Ss[m][nB] = acc1[v] * scale + bias2d[((size_t)Hh * 64 + m) * 64 + nB];
  }
  __syncthreads();

  if (tid < 64) {
    float mx = -1e30f;
#pragma unroll 4
    for (int j = 0; j < 64; ++j) mx = fmaxf(mx, Ss[tid][j]);
    float sum = 0.f;
#pragma unroll 4
    for (int j = 0; j < 64; ++j) { float e = __expf(Ss[tid][j] - mx); Ss[tid][j] = e; sum += e; }
    float inv = 1.0f / sum;
    size_t o = (size_t)bH * 4096 + (size_t)tid * 64;
#pragma unroll 4
    for (int j = 0; j < 64; ++j) attn[o + j] = Ss[tid][j] * inv;
  }
}

// ---------------------------------------------------------------------------
// Stage A (per b,H): r[64,2048] = attn_h[64,64] . Vmat[64,2048]
// Vmat[j, w*32+dv] = vlin[(b*4096 + j*64 + w)*256 + Hh*32 + dv]
// ---------------------------------------------------------------------------
__global__ __launch_bounds__(256) void stageA_kernel(
    const float* __restrict__ attn_h, const _Float16* __restrict__ vlin,
    _Float16* __restrict__ rbuf) {
  __shared__ __align__(16) _Float16 As[64 * 72];   // attn, K=64
  __shared__ __align__(16) _Float16 Bs[64 * 40];
  const int tid = threadIdx.x, lane = tid & 31, wv = tid >> 5;
  const int bH = blockIdx.y, b = bH >> 3, Hh = bH & 7;
  const int n0 = blockIdx.x * 64;            // over 2048 = w*32+dv
  const int tm = (wv & 3) * 16, tn = (wv >> 2) * 32;
  const int h = lane >> 4, l15 = lane & 15;
  const int w0 = n0 >> 5;                    // covers w0, w0+1

#pragma unroll
  for (int i = 0; i < 2; ++i) {              // attn 64x64 = 512 x 8-half chunks
    int p = tid + 256 * i;
    int r = p >> 3, co = p & 7;
    *(v8h*)&As[r * 72 + 8 * co] = load8h(&attn_h[(size_t)bH * 4096 + r * 64 + 8 * co]);
  }

  v8f acc0 = {}, acc1 = {};
  for (int kk = 0; kk < 64; kk += 32) {
#pragma unroll
    for (int i = 0; i < 4; ++i) {            // B: pairs along n (dv-contig)
      int p = tid + 256 * i;
      int k = p >> 5, np = p & 31;
      int n = 2 * np, w = w0 + (n >> 5), dv = n & 31;
      v2h g = load2h(&vlin[((size_t)b * 4096 + (kk + k) * 64 + w) * 256 + Hh * 32 + dv]);
      Bs[n * 40 + k]       = g[0];
      Bs[(n + 1) * 40 + k] = g[1];
    }
    __syncthreads();
    v16h af  = afrag(&As[(tm + l15) * 72 + kk], h);
    v16h bf0 = bfrag(&Bs[(tn + l15) * 40], h);
    v16h bf1 = bfrag(&Bs[(tn + 16 + l15) * 40], h);
    acc0 = wmma_f16(af, bf0, acc0);
    acc1 = wmma_f16(af, bf1, acc1);
    __syncthreads();
  }
  const int nA = n0 + tn + l15, nB = nA + 16;
#pragma unroll
  for (int v = 0; v < 8; ++v) {
    int m = tm + v + 8 * h;
    size_t o = ((size_t)bH * 64 + m) * 2048;
    rbuf[o + nA] = (_Float16)acc0[v];
    rbuf[o + nB] = (_Float16)acc1[v];
  }
}

// ---------------------------------------------------------------------------
// Stage B (per b,H): X[2048,64] . attn_w[64,64], rows m = i*32+dv.
// Scatter to R[(b*4096 + i*64 + w')*256 + Hh*32 + dv] (f16).
// ---------------------------------------------------------------------------
__global__ __launch_bounds__(256) void stageB_kernel(
    const _Float16* __restrict__ rbuf, const float* __restrict__ attn_w,
    _Float16* __restrict__ Rh) {
  __shared__ __align__(16) _Float16 As[64 * 72];   // A[m_local][k], K=64
  __shared__ __align__(16) _Float16 Bs[64 * 72];   // Bs[n][k] = attn_w[k*64+n]
  const int tid = threadIdx.x, lane = tid & 31, wv = tid >> 5;
  const int bH = blockIdx.y, b = bH >> 3, Hh = bH & 7;
  const int m0 = blockIdx.x * 64;
  const int tm = (wv & 3) * 16, tn = (wv >> 2) * 32;
  const int h = lane >> 4, l15 = lane & 15;
  const int i0 = m0 >> 5;                    // covers i0, i0+1

#pragma unroll
  for (int i = 0; i < 8; ++i) {              // A: pairs along m (dv-contig)
    int p = tid + 256 * i;
    int k = p >> 5, mp = p & 31;
    int ml = 2 * mp, mg = m0 + ml;
    int ii = i0 + (ml >> 5), dv = mg & 31;
    v2h g = load2h(&rbuf[((size_t)bH * 64 + ii) * 2048 + k * 32 + dv]);
    As[ml * 72 + k]       = g[0];
    As[(ml + 1) * 72 + k] = g[1];
  }
#pragma unroll
  for (int i = 0; i < 8; ++i) {              // B: pairs along n
    int p = tid + 256 * i;
    int k = p >> 5, np = p & 31;
    int n = 2 * np;
    v2h g = load2h(&attn_w[(size_t)bH * 4096 + k * 64 + n]);
    Bs[n * 72 + k]       = g[0];
    Bs[(n + 1) * 72 + k] = g[1];
  }
  __syncthreads();

  v8f acc0 = {}, acc1 = {};
#pragma unroll
  for (int kk = 0; kk < 64; kk += 32) {
    v16h af  = afrag(&As[(tm + l15) * 72 + kk], h);
    v16h bf0 = bfrag(&Bs[(tn + l15) * 72 + kk], h);
    v16h bf1 = bfrag(&Bs[(tn + 16 + l15) * 72 + kk], h);
    acc0 = wmma_f16(af, bf0, acc0);
    acc1 = wmma_f16(af, bf1, acc1);
  }
  const int nA = tn + l15, nB = nA + 16;
#pragma unroll
  for (int v = 0; v < 8; ++v) {
    int ml = m0 + tm + v + 8 * h;
    int ii = ml >> 5, dv = ml & 31;
    size_t base = ((size_t)b * 4096 + ii * 64);
    Rh[(base + nA) * 256 + Hh * 32 + dv] = (_Float16)acc0[v];
    Rh[(base + nB) * 256 + Hh * 32 + dv] = (_Float16)acc1[v];
  }
}

// ---------------------------------------------------------------------------
extern "C" void kernel_launch(void* const* d_in, const int* in_sizes, int n_in,
                              void* d_out, int out_size, void* d_ws, size_t ws_size,
                              hipStream_t stream) {
  (void)in_sizes; (void)n_in; (void)out_size; (void)ws_size;
  const float* x       = (const float*)d_in[0];
  const float* dw_qh_w = (const float*)d_in[1];
  const float* dw_qh_b = (const float*)d_in[2];
  const float* fc_qh_w = (const float*)d_in[3];
  const float* fc_qh_b = (const float*)d_in[4];
  const float* dw_kh_w = (const float*)d_in[5];
  const float* dw_kh_b = (const float*)d_in[6];
  const float* fc_kh_w = (const float*)d_in[7];
  const float* fc_kh_b = (const float*)d_in[8];
  const float* Bh      = (const float*)d_in[9];
  const float* dw_v_w  = (const float*)d_in[10];
  const float* dw_v_b  = (const float*)d_in[11];
  const float* fc_v_w  = (const float*)d_in[12];
  const float* fc_v_b  = (const float*)d_in[13];
  const float* dw_qw_w = (const float*)d_in[14];
  const float* dw_qw_b = (const float*)d_in[15];
  const float* fc_qw_w = (const float*)d_in[16];
  const float* fc_qw_b = (const float*)d_in[17];
  const float* dw_kw_w = (const float*)d_in[18];
  const float* dw_kw_b = (const float*)d_in[19];
  const float* fc_kw_w = (const float*)d_in[20];
  const float* fc_kw_b = (const float*)d_in[21];
  const float* Bw      = (const float*)d_in[22];
  const float* fc_o_w  = (const float*)d_in[23];
  const float* fc_o_b  = (const float*)d_in[24];

  char* ws = (char*)d_ws;
  size_t off = 0;
  _Float16* vconv = (_Float16*)(ws + off); off += (size_t)16 * 256 * 4096 * 2; // 32 MB
  float* qh_pool = (float*)(ws + off); off += 1024 * 256 * 4;
  float* kh_pool = (float*)(ws + off); off += 1024 * 256 * 4;
  float* qw_pool = (float*)(ws + off); off += 1024 * 256 * 4;
  float* kw_pool = (float*)(ws + off); off += 1024 * 256 * 4;
  float* qh_lin = (float*)(ws + off); off += 1024 * 256 * 4;
  float* kh_lin = (float*)(ws + off); off += 1024 * 256 * 4;
  float* qw_lin = (float*)(ws + off); off += 1024 * 256 * 4;
  float* kw_lin = (float*)(ws + off); off += 1024 * 256 * 4;
  float* attn_h = (float*)(ws + off); off += 128 * 4096 * 4;                   // 2 MB
  float* attn_w = (float*)(ws + off); off += 128 * 4096 * 4;                   // 2 MB
  _Float16* vlin = (_Float16*)(ws + off); off += (size_t)65536 * 256 * 2;      // 32 MB
  _Float16* Rh   = (_Float16*)(ws + off); off += (size_t)65536 * 256 * 2;      // 32 MB
  _Float16* rbuf = (_Float16*)vconv;   // alias: vconv dead after fc_v GEMM

  const dim3 blk(256);
  const float scale = 0.17677669529663687f;  // 32^-0.5

  dwconv_pool_kernel<<<dim3(16 * 256), blk, 0, stream>>>(
      x, dw_qh_w, dw_qh_b, dw_kh_w, dw_kh_b, dw_v_w, dw_v_b,
      dw_qw_w, dw_qw_b, dw_kw_w, dw_kw_b,
      vconv, qh_pool, kh_pool, qw_pool, kw_pool);

  gemm_lin_kernel<float, float><<<dim3(4, 8), blk, 0, stream>>>(
      qh_pool, fc_qh_w, fc_qh_b, qh_lin, 1024, 256, 256);
  gemm_lin_kernel<float, float><<<dim3(4, 8), blk, 0, stream>>>(
      kh_pool, fc_kh_w, fc_kh_b, kh_lin, 1024, 256, 256);
  gemm_lin_kernel<float, float><<<dim3(4, 8), blk, 0, stream>>>(
      qw_pool, fc_qw_w, fc_qw_b, qw_lin, 1024, 256, 256);
  gemm_lin_kernel<float, float><<<dim3(4, 8), blk, 0, stream>>>(
      kw_pool, fc_kw_w, fc_kw_b, kw_lin, 1024, 256, 256);

  attn_kernel<<<dim3(128), blk, 0, stream>>>(qh_lin, kh_lin, Bh, attn_h, scale);
  attn_kernel<<<dim3(128), blk, 0, stream>>>(qw_lin, kw_lin, Bw, attn_w, scale);

  gemm_fcv_kernel<<<dim3(4, 512), blk, 0, stream>>>(vconv, fc_v_w, fc_v_b, vlin);

  stageA_kernel<<<dim3(32, 128), blk, 0, stream>>>(attn_h, vlin, rbuf);
  stageB_kernel<<<dim3(32, 128), blk, 0, stream>>>(rbuf, attn_w, Rh);

  gemm_lin_kernel<_Float16, float><<<dim3(4, 512), blk, 0, stream>>>(
      Rh, fc_o_w, fc_o_b, (float*)d_out, 65536, 256, 256);
}
